// GNNEncoder_59339268162112
// MI455X (gfx1250) — matrix operand order repaired and checked by reading, not compiled, not run on hardware
//
#include <hip/hip_runtime.h>
#include <hip/hip_bf16.h>
#include <math.h>

#define NEG_SLOPE 0.2f
#define EPS_F 1e-16f

typedef __attribute__((ext_vector_type(16))) __bf16 v16bf;
typedef __attribute__((ext_vector_type(8)))  float  v8f;

// ---------- small utility kernels ----------

__global__ void zero_f32_kernel(float* __restrict__ p, long long n) {
    long long i = (long long)blockIdx.x * blockDim.x + threadIdx.x;
    if (i < n) p[i] = 0.0f;
}

__global__ void f32_to_bf16_kernel(const float* __restrict__ src,
                                   __bf16* __restrict__ dst, long long n) {
    long long i = (long long)blockIdx.x * blockDim.x + threadIdx.x;
    if (i < n) dst[i] = (__bf16)src[i];
}

// src is [K][M] row-major f32; dst is [M][K] row-major bf16 (transposed pack)
__global__ void transpose_to_bf16_kernel(const float* __restrict__ src,
                                         __bf16* __restrict__ dst,
                                         int K, int M) {
    long long i = (long long)blockIdx.x * blockDim.x + threadIdx.x;
    long long n = (long long)K * M;
    if (i >= n) return;
    int m = (int)(i / K);
    int k = (int)(i - (long long)m * K);
    dst[i] = (__bf16)src[(long long)k * M + m];
}

// out[i] += bias[i % C]
__global__ void add_bias_kernel(float* __restrict__ out, const float* __restrict__ bias,
                                int C, long long n) {
    long long i = (long long)blockIdx.x * blockDim.x + threadIdx.x;
    if (i < n) out[i] += bias[(int)(i % C)];
}

// hbf[i] = bf16( elu( agg[i] + bias[i % C] ) )
__global__ void finalize_elu_bf16_kernel(const float* __restrict__ agg,
                                         const float* __restrict__ bias,
                                         __bf16* __restrict__ hbf,
                                         int C, long long n) {
    long long i = (long long)blockIdx.x * blockDim.x + threadIdx.x;
    if (i >= n) return;
    float v = agg[i] + bias[(int)(i % C)];
    v = (v > 0.0f) ? v : (__expf(v) - 1.0f);
    hbf[i] = (__bf16)v;
}

// ---------- WMMA GEMM: C[N,M] (f32) = A[N,K] (bf16) * Wt[M,K]^T (bf16) ----------
// One wave computes one 16x16 tile of C. blockDim.x = 128 (4 waves -> 4 col tiles).
__global__ void gemm_bf16_wmma_kernel(const __bf16* __restrict__ A,   // [N][K]
                                      const __bf16* __restrict__ Wt,  // [M][K]
                                      float* __restrict__ C,          // [N][M]
                                      int Nrows, int K, int M) {
    const int lane = threadIdx.x & 31;
    const int wave = threadIdx.x >> 5;
    const int row0 = blockIdx.x << 4;
    const int col0 = ((blockIdx.y << 2) + wave) << 4;
    if (col0 >= M) return;

    // A operand layout (16-bit A, 16x32): lanes 0-15 -> M=lane, k in {0..7,16..23};
    // lanes 16-31 -> same M, k in {8..15,24..31}; 2 bf16 packed per VGPR along K.
    const int koffA = (lane >> 4) << 3;      // 0 or 8
    // B operand layout (16-bit B, 32x16): lane -> N (mod 16); lanes 0-15 -> k in {0..7,16..23};
    // lanes 16-31 -> k in {8..15,24..31}; 2 bf16 packed per VGPR along K.
    const int koffB = (lane >> 4) << 3;      // 0 or 8

    const __bf16* arow = A  + (size_t)(row0 + (lane & 15)) * K;
    const __bf16* brow = Wt + (size_t)(col0 + (lane & 15)) * K;

    v8f acc = {0.f, 0.f, 0.f, 0.f, 0.f, 0.f, 0.f, 0.f};

    for (int k0 = 0; k0 < K; k0 += 32) {
        union { v16bf v; unsigned int u[8]; } a, b;
#pragma unroll
        for (int j = 0; j < 8; ++j) {
            const int ka = k0 + ((j & 4) << 2) + koffA + ((j & 3) << 1);
            const int kb = k0 + ((j & 4) << 2) + koffB + ((j & 3) << 1);
            a.u[j] = *(const unsigned int*)(arow + ka);
            b.u[j] = *(const unsigned int*)(brow + kb);
        }
        acc = __builtin_amdgcn_wmma_f32_16x16x32_bf16(
            /*neg_a=*/false, a.v, /*neg_b=*/false, b.v,
            /*c_mod=*/(short)0, acc, /*reuse_a=*/false, /*reuse_b=*/false);
    }

    // C/D layout: lanes 0-15 -> N=lane, VGPR r -> M=r; lanes 16-31 -> M=r+8.
    const int mbase = (lane >> 4) << 3;
    float* crow = C + (size_t)(row0 + mbase) * M + col0 + (lane & 15);
#pragma unroll
    for (int r = 0; r < 8; ++r)
        crow[(size_t)r * M] = acc[r];
}

// ---------- edge-parallel attention ----------

// monotonic key encoding for float atomic max via atomicMax(uint)
__device__ inline unsigned int fmax_key(float v) {
    unsigned int u = __float_as_uint(v);
    return (u & 0x80000000u) ? ~u : (u | 0x80000000u);
}
__device__ inline float fmax_decode(unsigned int key) {
    unsigned int u = (key & 0x80000000u) ? (key & 0x7fffffffu) : ~key;
    return __uint_as_float(u);
}

// pass 1 (one wave per edge): lanes stripe the flattened [H][C] feature vector;
// per-head butterfly reduction over the 32/H-lane subgroup, head leader stores
// the score and atomic-maxes the destination row.
__global__ void edge_score_warp_kernel(const int* __restrict__ src, const int* __restrict__ dst,
                                       const float* __restrict__ xl, const float* __restrict__ xr,
                                       const float* __restrict__ att,
                                       float* __restrict__ escore,
                                       unsigned int* __restrict__ emaxKey,
                                       int E, int H, int C) {
    const int e = blockIdx.x * (blockDim.x >> 5) + (threadIdx.x >> 5);
    if (e >= E) return;
    const int lane = threadIdx.x & 31;
    const int per_lane = (H * C) >> 5;            // 8 (L1) or 4 (L2) floats per lane
    const int base = lane * per_lane;             // offset in flattened [H][C]
    const int h = base / C;                       // chunk never crosses a head
    const int s = src[e], d = dst[e];

    const float* pl = xl + (size_t)s * H * C + base;
    const float* pr = xr + (size_t)d * H * C + base;
    const float* pa = att + base;

    float acc = 0.0f;
    for (int t = 0; t < per_lane; t += 4) {
        float4 a = *(const float4*)(pl + t);
        float4 b = *(const float4*)(pr + t);
        float4 w = *(const float4*)(pa + t);
        float m;
        m = a.x + b.x; m = (m > 0.f) ? m : NEG_SLOPE * m; acc += m * w.x;
        m = a.y + b.y; m = (m > 0.f) ? m : NEG_SLOPE * m; acc += m * w.y;
        m = a.z + b.z; m = (m > 0.f) ? m : NEG_SLOPE * m; acc += m * w.z;
        m = a.w + b.w; m = (m > 0.f) ? m : NEG_SLOPE * m; acc += m * w.w;
    }

    const int lph = 32 / H;                       // lanes per head
    for (int off = lph >> 1; off > 0; off >>= 1)
        acc += __shfl_xor(acc, off, 32);

    if ((lane & (lph - 1)) == 0) {
        escore[(size_t)e * H + h] = acc;
        atomicMax(&emaxKey[(size_t)d * H + h], fmax_key(acc));
    }
}

// pass 2 (per (edge,head) thread): escore <- exp(e - emax[dst]); denom[dst] += escore
__global__ void edge_exp_kernel(const int* __restrict__ dst,
                                float* __restrict__ escore,
                                const unsigned int* __restrict__ emaxKey,
                                float* __restrict__ denom,
                                int E, int H) {
    int idx = blockIdx.x * blockDim.x + threadIdx.x;
    if (idx >= E * H) return;
    int e = idx / H, h = idx - e * H;
    int d = dst[e];
    float mx = fmax_decode(emaxKey[(size_t)d * H + h]);
    float v = __expf(escore[idx] - mx);
    escore[idx] = v;
    atomicAdd(&denom[(size_t)d * H + h], v);
}

// pass 3 (one wave per edge): coalesced gather of xl[src] and coalesced burst of
// atomic adds into the destination row: agg[dst,h,:] += alpha * xl[src,h,:]
__global__ void edge_aggregate_warp_kernel(const int* __restrict__ src, const int* __restrict__ dst,
                                           const float* __restrict__ xl,
                                           const float* __restrict__ escore,
                                           const float* __restrict__ denom,
                                           float* __restrict__ agg,
                                           int E, int H, int C) {
    const int e = blockIdx.x * (blockDim.x >> 5) + (threadIdx.x >> 5);
    if (e >= E) return;
    const int lane = threadIdx.x & 31;
    const int per_lane = (H * C) >> 5;
    const int base = lane * per_lane;
    const int h = base / C;
    const int s = src[e], d = dst[e];

    const float alpha = escore[(size_t)e * H + h] /
                        (denom[(size_t)d * H + h] + EPS_F);

    const float* pl = xl + (size_t)s * H * C + base;
    float* po = agg + (size_t)d * H * C + base;
    for (int t = 0; t < per_lane; t += 4) {
        float4 v = *(const float4*)(pl + t);
        atomicAdd(po + t + 0, alpha * v.x);
        atomicAdd(po + t + 1, alpha * v.y);
        atomicAdd(po + t + 2, alpha * v.z);
        atomicAdd(po + t + 3, alpha * v.w);
    }
}

// ---------- launch ----------

static inline int cdiv_i(long long n, int b) { return (int)((n + b - 1) / b); }

extern "C" void kernel_launch(void* const* d_in, const int* in_sizes, int n_in,
                              void* d_out, int out_size, void* d_ws, size_t ws_size,
                              hipStream_t stream) {
    (void)in_sizes; (void)n_in; (void)out_size; (void)ws_size;
    const int N = 50000, E = 800000;
    const int IN_CH = 128, HID = 64, HEADS = 4, OUT_CH = 128;
    const int F1 = HEADS * HID;  // 256

    const float* x    = (const float*)d_in[0];
    const int*   ei   = (const int*)  d_in[1];
    const float* Wl1  = (const float*)d_in[2];
    const float* Wr1  = (const float*)d_in[3];
    const float* att1 = (const float*)d_in[4];
    const float* b1   = (const float*)d_in[5];
    const float* Wl2  = (const float*)d_in[6];
    const float* Wr2  = (const float*)d_in[7];
    const float* att2 = (const float*)d_in[8];
    const float* b2   = (const float*)d_in[9];
    const int* src = ei;
    const int* dst = ei + E;
    float* out = (float*)d_out;

    // workspace bump allocator
    char* ws = (char*)d_ws;
    size_t off = 0;
    auto alloc = [&](size_t bytes) -> void* {
        void* p = ws + off;
        off += (bytes + 255) & ~(size_t)255;
        return p;
    };
    __bf16* wl1t = (__bf16*)alloc((size_t)IN_CH * F1 * 2);   // [256][128]
    __bf16* wr1t = (__bf16*)alloc((size_t)IN_CH * F1 * 2);
    __bf16* wl2t = (__bf16*)alloc((size_t)F1 * OUT_CH * 2);  // [128][256]
    __bf16* wr2t = (__bf16*)alloc((size_t)F1 * OUT_CH * 2);
    __bf16* xbf  = (__bf16*)alloc((size_t)N * IN_CH * 2);
    __bf16* hbf  = (__bf16*)alloc((size_t)N * F1 * 2);
    float*  xl   = (float*) alloc((size_t)N * F1 * 4);       // reused layer 2 (smaller)
    float*  xr   = (float*) alloc((size_t)N * F1 * 4);
    float*  agg1 = (float*) alloc((size_t)N * F1 * 4);
    float*  esc  = (float*) alloc((size_t)E * HEADS * 4);
    float*  emax = (float*) alloc((size_t)N * HEADS * 4);
    float*  den  = (float*) alloc((size_t)N * HEADS * 4);

    const int T = 256;               // 8 waves per block
    const int WPB = T >> 5;          // waves (edges) per block

    // pack inputs / weights to bf16
    f32_to_bf16_kernel<<<cdiv_i((long long)N * IN_CH, T), T, 0, stream>>>(x, xbf, (long long)N * IN_CH);
    transpose_to_bf16_kernel<<<cdiv_i((long long)IN_CH * F1, T), T, 0, stream>>>(Wl1, wl1t, IN_CH, F1);
    transpose_to_bf16_kernel<<<cdiv_i((long long)IN_CH * F1, T), T, 0, stream>>>(Wr1, wr1t, IN_CH, F1);
    transpose_to_bf16_kernel<<<cdiv_i((long long)F1 * OUT_CH, T), T, 0, stream>>>(Wl2, wl2t, F1, OUT_CH);
    transpose_to_bf16_kernel<<<cdiv_i((long long)F1 * OUT_CH, T), T, 0, stream>>>(Wr2, wr2t, F1, OUT_CH);

    // ---- layer 1 ----
    {
        dim3 grid(N / 16, F1 / 64);
        gemm_bf16_wmma_kernel<<<grid, 128, 0, stream>>>(xbf, wl1t, xl, N, IN_CH, F1);
        gemm_bf16_wmma_kernel<<<grid, 128, 0, stream>>>(xbf, wr1t, xr, N, IN_CH, F1);
    }
    zero_f32_kernel<<<cdiv_i((long long)N * HEADS, T), T, 0, stream>>>(emax, (long long)N * HEADS); // key 0 == -inf
    zero_f32_kernel<<<cdiv_i((long long)N * HEADS, T), T, 0, stream>>>(den,  (long long)N * HEADS);
    zero_f32_kernel<<<cdiv_i((long long)N * F1, T), T, 0, stream>>>(agg1, (long long)N * F1);

    edge_score_warp_kernel<<<cdiv_i(E, WPB), T, 0, stream>>>(
        src, dst, xl, xr, att1, esc, (unsigned int*)emax, E, HEADS, HID);
    edge_exp_kernel<<<cdiv_i((long long)E * HEADS, T), T, 0, stream>>>(
        dst, esc, (const unsigned int*)emax, den, E, HEADS);
    edge_aggregate_warp_kernel<<<cdiv_i(E, WPB), T, 0, stream>>>(
        src, dst, xl, esc, den, agg1, E, HEADS, HID);

    finalize_elu_bf16_kernel<<<cdiv_i((long long)N * F1, T), T, 0, stream>>>(
        agg1, b1, hbf, F1, (long long)N * F1);

    // ---- layer 2 (heads=1, concat=False -> mean over 1 head == identity) ----
    {
        dim3 grid(N / 16, OUT_CH / 64);
        gemm_bf16_wmma_kernel<<<grid, 128, 0, stream>>>(hbf, wl2t, xl, N, F1, OUT_CH);
        gemm_bf16_wmma_kernel<<<grid, 128, 0, stream>>>(hbf, wr2t, xr, N, F1, OUT_CH);
    }
    zero_f32_kernel<<<cdiv_i((long long)N, T), T, 0, stream>>>(emax, (long long)N);
    zero_f32_kernel<<<cdiv_i((long long)N, T), T, 0, stream>>>(den,  (long long)N);
    zero_f32_kernel<<<cdiv_i((long long)N * OUT_CH, T), T, 0, stream>>>(out, (long long)N * OUT_CH);

    edge_score_warp_kernel<<<cdiv_i(E, WPB), T, 0, stream>>>(
        src, dst, xl, xr, att2, esc, (unsigned int*)emax, E, 1, OUT_CH);
    edge_exp_kernel<<<cdiv_i(E, T), T, 0, stream>>>(
        dst, esc, (const unsigned int*)emax, den, E, 1);
    edge_aggregate_warp_kernel<<<cdiv_i(E, WPB), T, 0, stream>>>(
        src, dst, xl, esc, den, out, E, 1, OUT_CH);

    add_bias_kernel<<<cdiv_i((long long)N * OUT_CH, T), T, 0, stream>>>(
        out, b2, OUT_CH, (long long)N * OUT_CH);
}
